// MolSpaceGNN_87978110091593
// MI455X (gfx1250) — matrix operand order repaired and checked by reading, not compiled, not run on hardware
//
#include <hip/hip_runtime.h>
#include <cstdint>
#include <cstddef>

#define DEV __device__ __forceinline__

// ---------------------------------------------------------------------------
// Types for CDNA5 WMMA (wave32, 16x16x32 bf16 -> f32)
// ---------------------------------------------------------------------------
typedef __attribute__((ext_vector_type(16))) __bf16        v16bf;
typedef __attribute__((ext_vector_type(8)))  float         v8f;
typedef __attribute__((ext_vector_type(8)))  unsigned int  v8u;

static constexpr int D  = 128;
static constexpr int H  = 256;
static constexpr int NA = 20000;
static constexpr int EA = 320000;
static constexpr int EB = 320000;

// ---------------------------------------------------------------------------
// Helpers
// ---------------------------------------------------------------------------
DEV unsigned short f2bf(float f) {
  union { float f; unsigned int u; } cv; cv.f = f;
  unsigned int u = cv.u;
  u += 0x7FFFu + ((u >> 16) & 1u);   // round-to-nearest-even
  return (unsigned short)(u >> 16);
}

DEV float sigmoid_(float x) { return 1.0f / (1.0f + __expf(-x)); }
DEV float silu_(float x)    { return x * sigmoid_(x); }

DEV void atomAddF(float* p, float v) {
  __hip_atomic_fetch_add(p, v, __ATOMIC_RELAXED, __HIP_MEMORY_SCOPE_AGENT);
}

// A-matrix (16xK tile, bf16, row-major in LDS with row stride `rowStride`)
// CDNA5 16-bit A layout: lane L (M = L&15, half = L>>4) holds
//   VGPR0..3 : K = 8*half + {0..7}   (contiguous)
//   VGPR4..7 : K = 16 + 8*half + {0..7} (contiguous)
// -> two 16-byte LDS loads per fragment.
DEV v16bf load_a_frag(const unsigned short* lds, int rowStride, int kbase, int lane) {
  const int mrow = lane & 15;
  const int hf   = lane >> 4;
  const unsigned short* p = lds + mrow * rowStride + kbase + 8 * hf;
  uint4 a0 = *(const uint4*)(p);
  uint4 a1 = *(const uint4*)(p + 16);
  v8u u;
  u[0]=a0.x; u[1]=a0.y; u[2]=a0.z; u[3]=a0.w;
  u[4]=a1.x; u[5]=a1.y; u[6]=a1.z; u[7]=a1.w;
  return __builtin_bit_cast(v16bf, u);
}

// B-matrix fragments are pre-packed so that lane `l` of fragment `frag` owns
// 16 contiguous bf16 values (its column, K rows 8h+0..7 then 16+8h+0..7, h = l/16) -> 32B/lane.
DEV v16bf load_b_frag(const unsigned short* __restrict__ pk, int frag, int lane) {
  const uint4* p = (const uint4*)(pk + ((size_t)frag * 32 + (size_t)lane) * 16);
  uint4 b0 = p[0];
  uint4 b1 = p[1];
  v8u u;
  u[0]=b0.x; u[1]=b0.y; u[2]=b0.z; u[3]=b0.w;
  u[4]=b1.x; u[5]=b1.y; u[6]=b1.z; u[7]=b1.w;
  return __builtin_bit_cast(v16bf, u);
}

DEV v8f wmma_bf16(v16bf a, v16bf b, v8f c) {
  return __builtin_amdgcn_wmma_f32_16x16x32_bf16(
      /*neg_a=*/false, a, /*neg_b=*/false, b,
      /*c_mod=*/(short)0, c, /*reuse_a=*/false, /*reuse_b=*/false);
}

// ---------------------------------------------------------------------------
// Weight pack: f32 (K x N row-major) -> bf16 WMMA-B fragment order.
// frag f = kt*(N/16)+nt ; within fragment: lane l -> column nt*16+(l&15),
// rows kt*32 + ((l>=16)?8:0) + (j&7) + ((j>=8)?16:0), j=0..15 stored contiguously.
// ---------------------------------------------------------------------------
__global__ void pack_weight_kernel(const float* __restrict__ W,
                                   unsigned short* __restrict__ out,
                                   int K, int N) {
  const int total  = K * N;
  const int ntiles = N >> 4;
  for (int d = blockIdx.x * blockDim.x + threadIdx.x; d < total;
       d += gridDim.x * blockDim.x) {
    int f    = d >> 9;        // fragment (512 elements each)
    int r    = d & 511;
    int lane = r >> 4;
    int j    = r & 15;
    int kt   = f / ntiles;
    int nt   = f - kt * ntiles;
    int k    = kt * 32 + ((lane >= 16) ? 8 : 0) + (j & 7) + ((j >> 3) * 16);
    int n    = nt * 16 + (lane & 15);
    out[d] = f2bf(W[(size_t)k * N + n]);
  }
}

__global__ void zero_f32_kernel(float* __restrict__ p, long long n4) {
  float4 z = {0.f, 0.f, 0.f, 0.f};
  float4* p4 = (float4*)p;
  for (long long i = blockIdx.x * (long long)blockDim.x + threadIdx.x; i < n4;
       i += gridDim.x * (long long)blockDim.x)
    p4[i] = z;
}

// ---------------------------------------------------------------------------
// Edge gated-MLP kernel. One 16-edge tile per block, 128 threads = 4 waves.
//   x (16 x NSEG*128) = concat of gathered feature rows
//   hid = silu(x @ W1 + b1)  (both o- and g-paths)
//   y   = silu(hid_o @ W2o + b2o) * sigmoid(hid_g @ W2g + b2g)
// MODE 0 (atom):  y *= edge_weight[e];                scatter-add to out[scA[e]]
// MODE 1 (bond):  y *= nw[scA[e]]*nw[scB[e]];         scatter-add to out[scB[e]]
// MODE 2 (angle): out[e] = addBase[e] + y
// ---------------------------------------------------------------------------
template <int NSEG, int MODE>
__global__ __launch_bounds__(128) void edge_gated_mlp_kernel(
    const float* __restrict__ seg0, const int* __restrict__ idx0,
    const float* __restrict__ seg1, const int* __restrict__ idx1,
    const float* __restrict__ seg2, const int* __restrict__ idx2,
    const float* __restrict__ seg3, const int* __restrict__ idx3,
    int idx3Stride, int idx3Off,
    const unsigned short* __restrict__ pW1o, const float* __restrict__ b1o,
    const unsigned short* __restrict__ pW2o, const float* __restrict__ b2o,
    const unsigned short* __restrict__ pW1g, const float* __restrict__ b1g,
    const unsigned short* __restrict__ pW2g, const float* __restrict__ b2g,
    const float* __restrict__ wsrc,
    const int* __restrict__ scA, const int* __restrict__ scB,
    const float* __restrict__ addBase,
    float* __restrict__ outp)
{
  constexpr int KIN = NSEG * 128;
  __shared__ alignas(16) unsigned short xLds[16 * KIN];
  __shared__ alignas(16) unsigned short oLds[16 * 256];
  __shared__ alignas(16) unsigned short gLds[16 * 256];

  const int tid = threadIdx.x;
  const int blockBase = blockIdx.x * 16;

  // ---- stage gathered input tile as bf16 into LDS ----
  for (int i = tid; i < NSEG * 1024; i += 128) {
    int seg = i >> 10;
    int rem = i & 1023;
    int row = rem >> 6;
    int col = (rem & 63) * 2;
    int e   = blockBase + row;
    const float* sp; int ri;
    if (seg == 0)      { sp = seg0; ri = idx0 ? idx0[e] : e; }
    else if (seg == 1) { sp = seg1; ri = idx1 ? idx1[e] : e; }
    else if (seg == 2) { sp = seg2; ri = idx2 ? idx2[e] : e; }
    else               { sp = seg3; ri = idx3[(size_t)e * idx3Stride + idx3Off]; }
    const float2 v = *(const float2*)(sp + (size_t)ri * 128 + col);
    unsigned int pk = (unsigned int)f2bf(v.x) | ((unsigned int)f2bf(v.y) << 16);
    *(unsigned int*)(&xLds[row * KIN + seg * 128 + col]) = pk;
  }
  __syncthreads();

  const int lane = tid & 31;
  const int wave = tid >> 5;
  const int nloc = lane & 15;
  const int hf   = lane >> 4;

  // ---- GEMM1: x(16xKIN) @ W1(KINx256), o and g paths share the A fragment ----
  v8f zero = {};
  v8f co[4], cg[4];
#pragma unroll
  for (int t = 0; t < 4; ++t) { co[t] = zero; cg[t] = zero; }

  constexpr int KT1 = KIN / 32;
  for (int kt = 0; kt < KT1; ++kt) {
    v16bf a = load_a_frag(xLds, KIN, kt * 32, lane);
#pragma unroll
    for (int t = 0; t < 4; ++t) {
      int nt = wave * 4 + t;
      co[t] = wmma_bf16(a, load_b_frag(pW1o, kt * 16 + nt, lane), co[t]);
      cg[t] = wmma_bf16(a, load_b_frag(pW1g, kt * 16 + nt, lane), cg[t]);
    }
  }

  // bias + SiLU, write hidden activations back to LDS as bf16
#pragma unroll
  for (int t = 0; t < 4; ++t) {
    int n = (wave * 4 + t) * 16 + nloc;
    float bo = b1o[n], bg = b1g[n];
#pragma unroll
    for (int r = 0; r < 8; ++r) {
      int mrow = r + 8 * hf;
      oLds[mrow * 256 + n] = f2bf(silu_(co[t][r] + bo));
      gLds[mrow * 256 + n] = f2bf(silu_(cg[t][r] + bg));
    }
  }
  __syncthreads();

  // ---- GEMM2: hid(16x256) @ W2(256x128) for both paths ----
  v8f o2[2], g2[2];
#pragma unroll
  for (int t = 0; t < 2; ++t) { o2[t] = zero; g2[t] = zero; }
#pragma unroll
  for (int kt = 0; kt < 8; ++kt) {
    v16bf ao = load_a_frag(oLds, 256, kt * 32, lane);
    v16bf ag = load_a_frag(gLds, 256, kt * 32, lane);
#pragma unroll
    for (int t = 0; t < 2; ++t) {
      int nt = wave * 2 + t;
      o2[t] = wmma_bf16(ao, load_b_frag(pW2o, kt * 8 + nt, lane), o2[t]);
      g2[t] = wmma_bf16(ag, load_b_frag(pW2g, kt * 8 + nt, lane), g2[t]);
    }
  }

  // ---- gate, weight, scatter ----
#pragma unroll
  for (int t = 0; t < 2; ++t) {
    int col = (wave * 2 + t) * 16 + nloc;
    float bo2 = b2o[col], bg2 = b2g[col];
#pragma unroll
    for (int r = 0; r < 8; ++r) {
      int mrow = r + 8 * hf;
      int e    = blockBase + mrow;
      float val = silu_(o2[t][r] + bo2) * sigmoid_(g2[t][r] + bg2);
      if constexpr (MODE == 0) {
        float w = wsrc[(size_t)e * 128 + col];
        int dd  = scA[e];
        atomAddF(outp + (size_t)dd * 128 + col, val * w);
      } else if constexpr (MODE == 1) {
        int sA = scA[e], sB = scB[e];
        float w = wsrc[(size_t)sA * 128 + col] * wsrc[(size_t)sB * 128 + col];
        atomAddF(outp + (size_t)sB * 128 + col, val * w);
      } else {
        outp[(size_t)e * 128 + col] = addBase[(size_t)e * 128 + col] + val;
      }
    }
  }
}

// ---------------------------------------------------------------------------
// Residual linear: out = feat + h @ W(128x128) + bias ; 16 rows per block.
// ---------------------------------------------------------------------------
__global__ __launch_bounds__(128) void residual_linear_kernel(
    const float* __restrict__ h, const float* __restrict__ feat,
    const unsigned short* __restrict__ pW, const float* __restrict__ bias,
    float* __restrict__ outF)
{
  __shared__ alignas(16) unsigned short hLds[16 * 128];
  const int tid  = threadIdx.x;
  const int base = blockIdx.x * 16;
  for (int i = tid; i < 1024; i += 128) {
    int row = i >> 6;
    int col = (i & 63) * 2;
    const float2 v = *(const float2*)(h + (size_t)(base + row) * 128 + col);
    *(unsigned int*)(&hLds[row * 128 + col]) =
        (unsigned int)f2bf(v.x) | ((unsigned int)f2bf(v.y) << 16);
  }
  __syncthreads();

  const int lane = tid & 31, wave = tid >> 5;
  v8f zero = {};
  v8f c[2]; c[0] = zero; c[1] = zero;
#pragma unroll
  for (int kt = 0; kt < 4; ++kt) {
    v16bf a = load_a_frag(hLds, 128, kt * 32, lane);
#pragma unroll
    for (int t = 0; t < 2; ++t)
      c[t] = wmma_bf16(a, load_b_frag(pW, kt * 8 + wave * 2 + t, lane), c[t]);
  }
  const int nloc = lane & 15, hf = lane >> 4;
#pragma unroll
  for (int t = 0; t < 2; ++t) {
    int col = (wave * 2 + t) * 16 + nloc;
    float b = bias[col];
#pragma unroll
    for (int r = 0; r < 8; ++r) {
      int row = base + r + 8 * hf;
      outF[(size_t)row * 128 + col] = feat[(size_t)row * 128 + col] + c[t][r] + b;
    }
  }
}

// ---------------------------------------------------------------------------
// Host launch
// ---------------------------------------------------------------------------
struct GatedIdx { int oW1, ob1, oW2, ob2, gW1, gb1, gW2, gb2; };

static GatedIdx gated_sorted(int s)    { return {s+4, s+6, s+5, s+7, s+0, s+2, s+1, s+3}; }
static GatedIdx gated_insertion(int s) { return {s+0, s+1, s+2, s+3, s+4, s+5, s+6, s+7}; }

extern "C" void kernel_launch(void* const* d_in, const int* in_sizes, int n_in,
                              void* d_out, int out_size, void* d_ws, size_t ws_size,
                              hipStream_t stream) {
  (void)n_in; (void)out_size; (void)ws_size;

  // ---- figure out input flatten convention from sizes ----
  int af, bf, anf, ew, nw, alinW, alinb, blinW, blinb;
  int asrc, adst, bsrc, bdst, aidx;
  GatedIdx A, B, G;   // atom, bond, angle gated params

  if (in_sizes[0] == NA * D) {
    // top-level dict in insertion order
    af = 0; bf = 1; anf = 2; ew = 3; nw = 4;
    if (in_sizes[5] == 3 * D * H) {
      // params dict flattened in insertion order: atom, bond, angle, lin*
      A = gated_insertion(5); B = gated_insertion(13); G = gated_insertion(21);
      alinW = 29; alinb = 30; blinW = 31; blinb = 32;
    } else {
      // params dict flattened with sorted keys: angle, atom, atom_lin*, bond, bond_lin*
      G = gated_sorted(5); A = gated_sorted(13);
      alinW = 21; alinb = 22;
      B = gated_sorted(23);
      blinW = 31; blinb = 32;
    }
    asrc = 33; adst = 34; bsrc = 35; bdst = 36; aidx = 37;
  } else {
    // whole top-level dict flattened with sorted keys
    anf = 0; aidx = 1; adst = 2; af = 3; asrc = 4; bdst = 5; bf = 6; bsrc = 7;
    ew = 8; nw = 9;
    G = gated_sorted(10); A = gated_sorted(18);
    alinW = 26; alinb = 27;
    B = gated_sorted(28);
    blinW = 36; blinb = 37;
  }

  auto F = [&](int i) { return (const float*)d_in[i]; };
  auto I = [&](int i) { return (const int*)d_in[i]; };

  // ---- workspace layout ----
  size_t off = 0;
  auto alloc = [&](size_t bytes) {
    off = (off + 255) & ~(size_t)255;
    void* p = (char*)d_ws + off;
    off += bytes;
    return p;
  };
  float* h_atom = (float*)alloc((size_t)NA * D * 4);   // contiguous with h_bond
  float* h_bond = (float*)alloc((size_t)EA * D * 4);

  auto allocPk = [&](int K, int N) { return (unsigned short*)alloc((size_t)K * N * 2); };
  unsigned short* pA_oW1 = allocPk(3 * D, H);
  unsigned short* pA_gW1 = allocPk(3 * D, H);
  unsigned short* pA_oW2 = allocPk(H, D);
  unsigned short* pA_gW2 = allocPk(H, D);
  unsigned short* pB_oW1 = allocPk(4 * D, H);
  unsigned short* pB_gW1 = allocPk(4 * D, H);
  unsigned short* pB_oW2 = allocPk(H, D);
  unsigned short* pB_gW2 = allocPk(H, D);
  unsigned short* pG_oW1 = allocPk(4 * D, H);
  unsigned short* pG_gW1 = allocPk(4 * D, H);
  unsigned short* pG_oW2 = allocPk(H, D);
  unsigned short* pG_gW2 = allocPk(H, D);
  unsigned short* pAlin  = allocPk(D, D);
  unsigned short* pBlin  = allocPk(D, D);

  // ---- output sections ----
  float* atom_out  = (float*)d_out;
  float* bond_out  = atom_out + (size_t)NA * D;
  float* angle_out = bond_out + (size_t)EA * D;

  // ---- 0) zero segment-sum accumulators ----
  long long zn4 = ((long long)NA * D + (long long)EA * D) / 4;
  zero_f32_kernel<<<2048, 256, 0, stream>>>(h_atom, zn4);

  // ---- 1) pack weights to bf16 WMMA-B fragment order ----
  auto pack = [&](const float* W, unsigned short* dst, int K, int N) {
    int total = K * N;
    pack_weight_kernel<<<(total + 255) / 256, 256, 0, stream>>>(W, dst, K, N);
  };
  pack(F(A.oW1), pA_oW1, 3 * D, H);  pack(F(A.gW1), pA_gW1, 3 * D, H);
  pack(F(A.oW2), pA_oW2, H, D);      pack(F(A.gW2), pA_gW2, H, D);
  pack(F(B.oW1), pB_oW1, 4 * D, H);  pack(F(B.gW1), pB_gW1, 4 * D, H);
  pack(F(B.oW2), pB_oW2, H, D);      pack(F(B.gW2), pB_gW2, H, D);
  pack(F(G.oW1), pG_oW1, 4 * D, H);  pack(F(G.gW1), pG_gW1, 4 * D, H);
  pack(F(G.oW2), pG_oW2, H, D);      pack(F(G.gW2), pG_gW2, H, D);
  pack(F(alinW), pAlin, D, D);       pack(F(blinW), pBlin, D, D);

  // ---- 2) atom stage: x = [atom[src], atom[dst], bond[e]] ----
  edge_gated_mlp_kernel<3, 0><<<EA / 16, 128, 0, stream>>>(
      F(af), I(asrc), F(af), I(adst), F(bf), nullptr,
      nullptr, nullptr, 0, 0,
      pA_oW1, F(A.ob1), pA_oW2, F(A.ob2),
      pA_gW1, F(A.gb1), pA_gW2, F(A.gb2),
      F(ew), I(adst), nullptr, nullptr, h_atom);

  // ---- 3) atom_out = atom_feat + h_atom @ atom_linW + b ----
  residual_linear_kernel<<<NA / 16, 128, 0, stream>>>(
      h_atom, F(af), pAlin, F(alinb), atom_out);

  // ---- 4) bond stage: x = [bond[bsrc], bond[bdst], angle[e], atom_out[vtx]] ----
  edge_gated_mlp_kernel<4, 1><<<EB / 16, 128, 0, stream>>>(
      F(bf), I(bsrc), F(bf), I(bdst), F(anf), nullptr,
      atom_out, I(aidx), 3, 1,
      pB_oW1, F(B.ob1), pB_oW2, F(B.ob2),
      pB_gW1, F(B.gb1), pB_gW2, F(B.gb2),
      F(nw), I(bsrc), I(bdst), nullptr, h_bond);

  // ---- 5) bond_out = bond_feat + h_bond @ bond_linW + b ----
  residual_linear_kernel<<<EA / 16, 128, 0, stream>>>(
      h_bond, F(bf), pBlin, F(blinb), bond_out);

  // ---- 6) angle stage: x = [bond_out[bsrc], bond_out[bdst], angle, atom_out[vtx]] ----
  edge_gated_mlp_kernel<4, 2><<<EB / 16, 128, 0, stream>>>(
      bond_out, I(bsrc), bond_out, I(bdst), F(anf), nullptr,
      atom_out, I(aidx), 3, 1,
      pG_oW1, F(G.ob1), pG_oW2, F(G.ob2),
      pG_gW1, F(G.gb1), pG_gW2, F(G.gb2),
      nullptr, nullptr, nullptr, F(anf), angle_out);
}
